// SingleHeadSelfAttention_38010460569957
// MI455X (gfx1250) — matrix operand order repaired and checked by reading, not compiled, not run on hardware
//
#include <hip/hip_runtime.h>

#define EMB  768
#define EMBP 776          // padded xb row stride (bank-conflict-free LDS rows)
#define HS   64
#define BB   4
#define TT   4096
#define NT   (BB*TT)

typedef __attribute__((ext_vector_type(16))) __bf16 v16bf;
typedef __attribute__((ext_vector_type(8)))  __bf16 v8bf;
typedef __attribute__((ext_vector_type(4)))  __bf16 v4bf;
typedef __attribute__((ext_vector_type(8)))  float  v8f;
typedef __attribute__((ext_vector_type(4)))  float  v4f;
typedef __attribute__((ext_vector_type(4)))  unsigned int u32x4;
typedef __attribute__((ext_vector_type(8)))  int    i32x8;
typedef __attribute__((ext_vector_type(4)))  int    i32x4;

#define WMMA_BF16(a,b,c) \
  __builtin_amdgcn_wmma_f32_16x16x32_bf16(false,(a),false,(b),(short)0,(c),false,false)

static __device__ __forceinline__ v16bf cat16(v8bf lo, v8bf hi) {
  return __builtin_shufflevector(lo, hi, 0,1,2,3,4,5,6,7,8,9,10,11,12,13,14,15);
}
static __device__ __forceinline__ v8f vzero8() {
  v8f r;
#pragma unroll
  for (int i = 0; i < 8; ++i) r[i] = 0.0f;
  return r;
}
static __device__ __forceinline__ float halfmax(float v) {
  v = fmaxf(v, __shfl_xor(v, 1, 32));
  v = fmaxf(v, __shfl_xor(v, 2, 32));
  v = fmaxf(v, __shfl_xor(v, 4, 32));
  v = fmaxf(v, __shfl_xor(v, 8, 32));
  return v;
}
static __device__ __forceinline__ float halfsum(float v) {
  v += __shfl_xor(v, 1, 32);
  v += __shfl_xor(v, 2, 32);
  v += __shfl_xor(v, 4, 32);
  v += __shfl_xor(v, 8, 32);
  return v;
}

// ---------------- fp32 -> bf16 copy of x with padded row stride ----------------
__global__ __launch_bounds__(256) void cvt_x_kernel(const float* __restrict__ x,
                                                    __bf16* __restrict__ xb) {
  size_t i0 = ((size_t)blockIdx.x * 256 + threadIdx.x) * 8;   // over NT*EMB
  size_t row = i0 / EMB;
  int    col = (int)(i0 - row * EMB);                          // 8 | 768: chunk stays in row
  v4f a = *(const v4f*)(x + i0);
  v4f b = *(const v4f*)(x + i0 + 4);
  v4bf oa, ob;
#pragma unroll
  for (int i = 0; i < 4; ++i) { oa[i] = (__bf16)a[i]; ob[i] = (__bf16)b[i]; }
  __bf16* dst = xb + row * EMBP + col;
  *(v4bf*)(dst)     = oa;
  *(v4bf*)(dst + 4) = ob;
}

// ------- weights -> bf16, transposed: Wt[m*64+n][e] = W_m[e][n] -------
__global__ __launch_bounds__(256) void cvt_w_kernel(const float* __restrict__ Wk,
                                                    const float* __restrict__ Wq,
                                                    const float* __restrict__ Wv,
                                                    __bf16* __restrict__ Wt) {
  int idx = blockIdx.x * 256 + threadIdx.x;      // < 192*768
  int row = idx / EMB;
  int e   = idx - row * EMB;
  int m   = row >> 6, n = row & 63;
  const float* W = (m == 0) ? Wk : (m == 1) ? Wq : Wv;
  Wt[(size_t)row * EMB + e] = (__bf16)W[(size_t)e * HS + n];
}

// ---------------- projections: K, Q (row-major), V (transposed) ----------------
// grid 2048: strip = bx>>1 (16 rows), half = bx&1 selects 6 of 12 col-tiles.
// The 16x776 bf16 x-strip is DMA'd to LDS with one TDM descriptor.
__global__ __launch_bounds__(32) void proj_kernel(const __bf16* __restrict__ xb,
                                                  const __bf16* __restrict__ Wt,
                                                  const float* __restrict__ bk,
                                                  const float* __restrict__ bq,
                                                  const float* __restrict__ bv,
                                                  __bf16* __restrict__ Kb,
                                                  __bf16* __restrict__ Qb,
                                                  __bf16* __restrict__ Vt) {
  extern __shared__ __bf16 lx[];                 // 16 * 776 bf16 = 24832 B, offset 0
  const int lane = threadIdx.x;
  const int nl = lane & 15, hl = lane >> 4;
  const int r0    = (blockIdx.x >> 1) * 16;
  const int cbase = (blockIdx.x & 1) * 6;

  // --- Tensor Data Mover: contiguous 1-D tile, 16*EMBP bf16 elements ---
  {
    enum { NE = 16 * EMBP };                     // 12416 elements (2 B each)
    unsigned long long ga =
        (unsigned long long)(uintptr_t)(xb + (size_t)r0 * EMBP);
    u32x4 g0;
    g0[0] = 1u;                                  // count=1 (valid descriptor)
    g0[1] = 0u;                                  // lds_addr = 0 (dynamic LDS base)
    g0[2] = (unsigned)ga;                        // global_addr[31:0]
    g0[3] = (unsigned)(ga >> 32) | 0x80000000u;  // global_addr[56:32] | type=2
    i32x8 g1;
    g1[0] = (int)(1u << 16);                     // wg_mask=0, data_size=1 (2B)
    g1[1] = (int)((unsigned)(NE & 0xFFFF) << 16);// tensor_dim0[15:0]
    g1[2] = (int)(1u << 16);                     // tensor_dim0[31:16]=0 | tensor_dim1=1
    g1[3] = (int)((unsigned)(NE & 0xFFFF) << 16);// tensor_dim1 hi=0 | tile_dim0
    g1[4] = (int)1u;                             // tile_dim1=1, tile_dim2=0
    g1[5] = (int)NE;                             // tensor_dim0_stride[31:0]
    g1[6] = (int)((unsigned)(NE & 0xFFFF) << 16);// stride0 hi=0 | stride1[15:0]
    g1[7] = 0;                                   // stride1 hi
    i32x4 gz4;
    gz4[0] = gz4[1] = gz4[2] = gz4[3] = 0;
    i32x8 gz8;
#pragma unroll
    for (int i = 0; i < 8; ++i) gz8[i] = 0;
    __builtin_amdgcn_tensor_load_to_lds(g0, g1, gz4, gz4, gz8, 0);
    __builtin_amdgcn_s_wait_tensorcnt(0);
    asm volatile("" ::: "memory");
  }

  v8f acc[6];
#pragma unroll
  for (int i = 0; i < 6; ++i) acc[i] = vzero8();

  for (int k0 = 0; k0 < EMB; k0 += 32) {
    // A fragment from LDS: row stride 1552 B -> lane n starts at bank 4n (conflict-free)
    const __bf16* ab = lx + nl * EMBP + k0 + hl * 16;
    v16bf afrag = cat16(*(const v8bf*)ab, *(const v8bf*)(ab + 8));
#pragma unroll
    for (int i = 0; i < 6; ++i) {
      int ct = cbase + i;
      v16bf bfrag = *(const v16bf*)(Wt + (size_t)(ct * 16 + nl) * EMB + k0 + hl * 16);
      acc[i] = WMMA_BF16(afrag, bfrag, acc[i]);
    }
  }

#pragma unroll
  for (int i = 0; i < 6; ++i) {
    int ct    = cbase + i;
    int colin = (ct & 3) * 16 + nl;
    float bias = (ct < 4 ? bk : (ct < 8 ? bq : bv))[colin];
#pragma unroll
    for (int j = 0; j < 8; ++j) {
      float v = acc[i][j] + bias;
      int rg  = r0 + j + 8 * hl;
      if (ct < 4) {
        Kb[(size_t)rg * HS + colin] = (__bf16)v;
      } else if (ct < 8) {
        Qb[(size_t)rg * HS + colin] = (__bf16)v;
      } else {
        int b = rg >> 12, t = rg & (TT - 1);
        Vt[((size_t)b * HS + colin) * TT + t] = (__bf16)v;
      }
    }
  }
}

// ---------------- flash attention: one wave per (batch, 16-row t strip) ----------------
// score(t,s) = k_t . q_s (reference: k on left, no 1/sqrt scale), causal s<=t.
// Software-pipelined: block i issues block i+1's Q/V fragment loads before softmax.
__global__ __launch_bounds__(32) void attn_kernel(const __bf16* __restrict__ Kb,
                                                  const __bf16* __restrict__ Qb,
                                                  const __bf16* __restrict__ Vt,
                                                  float* __restrict__ out) {
  __shared__ __bf16 ps[16 * 32];   // wave-private P tile (row-major 16x32)

  const int lane = threadIdx.x;
  const int nl = lane & 15, hl = lane >> 4;
  const int tb  = blockIdx.x & (TT / 16 - 1);
  const int b   = blockIdx.x >> 8;
  const int tt0 = tb * 16;

  const __bf16* Kbb = Kb + (size_t)b * TT * HS;
  const __bf16* Qbb = Qb + (size_t)b * TT * HS;
  const __bf16* Vtb = Vt + (size_t)b * HS * TT;

  v8f acc[4];
#pragma unroll
  for (int i = 0; i < 4; ++i) acc[i] = vzero8();
  float m[8], l[8];
#pragma unroll
  for (int j = 0; j < 8; ++j) { m[j] = -1e30f; l[j] = 0.0f; }

  // A fragment of the K strip is loop-invariant (16 rows x 64 of HS)
  v16bf ka[2];
#pragma unroll
  for (int kk = 0; kk < 2; ++kk) {
    const __bf16* base = Kbb + (size_t)(tt0 + nl) * HS + kk * 32 + hl * 16;
    ka[kk] = cat16(*(const v8bf*)base, *(const v8bf*)(base + 8));
  }

  auto ldq = [&](int s0, v16bf q0[2], v16bf q1[2]) {
#pragma unroll
    for (int kk = 0; kk < 2; ++kk) {
      q0[kk] = *(const v16bf*)(Qbb + (size_t)(s0 + nl) * HS + kk * 32 + hl * 16);
      q1[kk] = *(const v16bf*)(Qbb + (size_t)(s0 + 16 + nl) * HS + kk * 32 + hl * 16);
    }
  };
  auto ldv = [&](int s0, v16bf vb[4]) {
#pragma unroll
    for (int hc = 0; hc < 4; ++hc)
      vb[hc] = *(const v16bf*)(Vtb + (size_t)(hc * 16 + nl) * TT + s0 + hl * 16);
  };

  const int sbmax = (tt0 + 15) >> 5;
  v16bf q0[2], q1[2], vb[4];
  ldq(0, q0, q1);
  ldv(0, vb);

  for (int sb = 0; sb <= sbmax; ++sb) {
    const int s0 = sb * 32;
    const bool more = (sb < sbmax);

    // S = K_strip (16x64) x Q_block^T (64x32)  -> two 16x16 tiles
    v8f S0 = vzero8(), S1 = vzero8();
    S0 = WMMA_BF16(ka[0], q0[0], S0);
    S1 = WMMA_BF16(ka[0], q1[0], S1);
    S0 = WMMA_BF16(ka[1], q0[1], S0);
    S1 = WMMA_BF16(ka[1], q1[1], S1);

    // prefetch next block's fragments; latency overlaps the softmax VALU work
    v16bf nq0[2], nq1[2], nvb[4];
    if (more) { ldq(s0 + 32, nq0, nq1); ldv(s0 + 32, nvb); }

    // causal mask (only the diagonal blocks need it)
    if (s0 + 31 > tt0) {
#pragma unroll
      for (int j = 0; j < 8; ++j) {
        int t = tt0 + j + 8 * hl;
        if (s0 + nl      > t) S0[j] = -1e30f;
        if (s0 + 16 + nl > t) S1[j] = -1e30f;
      }
    }

    // online softmax update
    float alpha[8];
#pragma unroll
    for (int j = 0; j < 8; ++j) {
      float mx = halfmax(fmaxf(S0[j], S1[j]));
      float mn = fmaxf(m[j], mx);
      float al = __expf(m[j] - mn);
      m[j] = mn;
      float p0 = __expf(S0[j] - mn);
      float p1 = __expf(S1[j] - mn);
      S0[j] = p0; S1[j] = p1;
      l[j] = l[j] * al + halfsum(p0 + p1);
      alpha[j] = al;
    }
#pragma unroll
    for (int hc = 0; hc < 4; ++hc)
#pragma unroll
      for (int j = 0; j < 8; ++j) acc[hc][j] *= alpha[j];

    // P (C-layout f32) -> bf16 row-major in LDS -> A fragment (16x32)
#pragma unroll
    for (int j = 0; j < 8; ++j) {
      int r = j + 8 * hl;
      ps[r * 32 + nl]      = (__bf16)S0[j];
      ps[r * 32 + 16 + nl] = (__bf16)S1[j];
    }
    asm volatile("s_wait_dscnt 0" ::: "memory");  // same-wave DS ordering, explicit
    v16bf pa = cat16(*(const v8bf*)(ps + nl * 32 + hl * 16),
                     *(const v8bf*)(ps + nl * 32 + hl * 16 + 8));

    // O += P (16x32) x V_block (32x64); V fragments were prefetched last iteration
#pragma unroll
    for (int hc = 0; hc < 4; ++hc)
      acc[hc] = WMMA_BF16(pa, vb[hc], acc[hc]);

    if (more) {
#pragma unroll
      for (int kk = 0; kk < 2; ++kk) { q0[kk] = nq0[kk]; q1[kk] = nq1[kk]; }
#pragma unroll
      for (int hc = 0; hc < 4; ++hc) vb[hc] = nvb[hc];
    }
  }

#pragma unroll
  for (int j = 0; j < 8; ++j) l[j] = 1.0f / l[j];
#pragma unroll
  for (int hc = 0; hc < 4; ++hc)
#pragma unroll
    for (int j = 0; j < 8; ++j) {
      int t = tt0 + j + 8 * hl;
      int h = hc * 16 + nl;
      out[((size_t)b * TT + t) * HS + h] = acc[hc][j] * l[j];
    }
}

extern "C" void kernel_launch(void* const* d_in, const int* in_sizes, int n_in,
                              void* d_out, int out_size, void* d_ws, size_t ws_size,
                              hipStream_t stream) {
  (void)in_sizes; (void)n_in; (void)out_size; (void)ws_size;
  const float* x  = (const float*)d_in[0];
  const float* Wk = (const float*)d_in[1];
  const float* bk = (const float*)d_in[2];
  const float* Wq = (const float*)d_in[3];
  const float* bq = (const float*)d_in[4];
  const float* Wv = (const float*)d_in[5];
  const float* bv = (const float*)d_in[6];

  // workspace layout (bf16): Kb | Qb | Vt | xb(padded) | Wt   (~30.6 MiB total)
  __bf16* Kb = (__bf16*)d_ws;
  __bf16* Qb = Kb + (size_t)NT * HS;
  __bf16* Vt = Qb + (size_t)NT * HS;
  __bf16* xb = Vt + (size_t)NT * HS;
  __bf16* Wt = xb + (size_t)NT * EMBP;

  cvt_x_kernel<<<(NT * EMB) / 2048, 256, 0, stream>>>(x, xb);
  cvt_w_kernel<<<(3 * HS * EMB) / 256, 256, 0, stream>>>(Wk, Wq, Wv, Wt);
  proj_kernel<<<(NT / 16) * 2, 32, 16 * EMBP * sizeof(__bf16), stream>>>(
      xb, Wt, bk, bq, bv, Kb, Qb, Vt);
  attn_kernel<<<BB * (TT / 16), 32, 0, stream>>>(Kb, Qb, Vt, (float*)d_out);
}
